// GRU_7241314861593
// MI455X (gfx1250) — compile-verified, matrix operand order checked by reading, not run on hardware
//
#include <hip/hip_runtime.h>
#include <stdint.h>

// Problem constants (match reference)
constexpr int kS  = 512;
constexpr int kB  = 64;
constexpr int kIN = 512;
constexpr int kH  = 1024;
constexpr int kKC = kIN + kH;      // 1536 concat K
constexpr int kNWG = 64;           // persistent WGs: 16 hidden units each
constexpr int kNT  = 384;          // 12 waves; 1 WMMA unit per wave
constexpr int kWB  = 3120;         // LDS W row stride bytes: 3072 + 3*16B TDM pad
constexpr int kHB  = 2064;         // LDS h row stride bytes: 2048 + 16B pad
// Dynamic LDS layout (bytes)
constexpr int LDS_W_OFF = 0;                        // 48 * 3120   = 149,760
constexpr int LDS_H_OFF = 48 * kWB;                 // + 64 * 2064 = 132,096
constexpr int LDS_S_OFF = LDS_H_OFF + kB * kHB;     // + 4*64*18*4 =  18,432
constexpr int LDS_TOTAL = LDS_S_OFF + 4 * kB * 18 * 4;  // 300,288

typedef __attribute__((ext_vector_type(16))) __bf16 v16bf;
typedef __attribute__((ext_vector_type(8)))  float  v8f;
typedef __attribute__((ext_vector_type(4)))  unsigned int v4u;
typedef __attribute__((ext_vector_type(8)))  int v8i;
typedef __attribute__((ext_vector_type(4)))  int v4i;

union FragBF { v16bf v; uint4 q[2]; };

__device__ __forceinline__ uint16_t f2bf(float x) {
    uint32_t u = __float_as_uint(x);
    u += 0x7FFFu + ((u >> 16) & 1u);
    return (uint16_t)(u >> 16);
}

// ---------------- workspace layout (bytes) ----------------
constexpr size_t OFF_XB   = 0;
constexpr size_t SZ_XB    = (size_t)kS * kB * kIN * 2;
constexpr size_t OFF_WCAT = OFF_XB + SZ_XB;
constexpr size_t SZ_WCAT  = (size_t)3 * kH * kKC * 2;
constexpr size_t OFF_HBF  = OFF_WCAT + SZ_WCAT;
constexpr size_t SZ_HBF   = (size_t)2 * kB * kH * 2;
constexpr size_t OFF_H32  = OFF_HBF + SZ_HBF;
constexpr size_t SZ_H32   = (size_t)2 * kB * kH * 4;
constexpr size_t OFF_CNT  = OFF_H32 + SZ_H32;

// ---------------- prep kernels ----------------
__global__ void k_cvt_x(const float* __restrict__ x, uint16_t* __restrict__ xb, int n) {
    for (int i = blockIdx.x * blockDim.x + threadIdx.x; i < n; i += gridDim.x * blockDim.x)
        xb[i] = f2bf(x[i]);
}

__global__ void k_build_w(const float* __restrict__ Wih, const float* __restrict__ Whh,
                          uint16_t* __restrict__ Wcat) {
    const int n = 3 * kH * kKC;
    for (int i = blockIdx.x * blockDim.x + threadIdx.x; i < n; i += gridDim.x * blockDim.x) {
        int j = i / kKC, k = i - j * kKC;
        float w = (k < kIN) ? Wih[(size_t)j * kIN + k] : Whh[(size_t)j * kH + (k - kIN)];
        Wcat[i] = f2bf(w);
    }
}

__global__ void k_init(uint16_t* __restrict__ hbf, float* __restrict__ h32, int* __restrict__ cnt) {
    int i = blockIdx.x * blockDim.x + threadIdx.x;
    if (i < kB * kH) { hbf[i] = 0; h32[i] = 0.0f; }
    if (i == 0) *cnt = 0;
}

// ---------------- TDM 2D bf16 tile load (ISA cdna5 §8 descriptor layout) ----------------
__device__ __forceinline__ void tdm_load_2d_bf16(uint32_t lds_byte_off, const void* gsrc,
                                                 uint32_t rows, uint32_t row_elems,
                                                 uint32_t pad_int_code, uint32_t pad_amt_code) {
#if __has_builtin(__builtin_amdgcn_tensor_load_to_lds)
    uint64_t ga = (uint64_t)(uintptr_t)gsrc;
    v4u g0;
    g0[0] = 1u;                                                // count=1, user mode
    g0[1] = lds_byte_off;                                      // lds_addr
    g0[2] = (uint32_t)ga;                                      // global_addr[31:0]
    g0[3] = ((uint32_t)(ga >> 32) & 0x01FFFFFFu) | (2u << 30); // global_addr[56:32] | type=2
    uint32_t d0 = (1u << 16);                                  // data_size = 2B
    if (pad_amt_code | pad_int_code)
        d0 |= (1u << 20) | (pad_int_code << 22) | (pad_amt_code << 25); // pad_enable
    v8i g1;
    g1[0] = (int)d0;                                                         // wg_mask=0 | flags
    g1[1] = (int)((row_elems & 0xFFFFu) << 16);                              // tensor_dim0 lo16
    g1[2] = (int)(((row_elems >> 16) & 0xFFFFu) | ((rows & 0xFFFFu) << 16)); // dim0 hi | dim1 lo
    g1[3] = (int)(((rows >> 16) & 0xFFFFu) | ((row_elems & 0xFFFFu) << 16)); // dim1 hi | tile_dim0
    g1[4] = (int)(rows & 0xFFFFu);                                           // tile_dim1 | tile_dim2=0
    g1[5] = (int)row_elems;                                                  // tensor_dim0_stride lo32
    g1[6] = 0;
    g1[7] = 0;
    v4i gz4 = {0, 0, 0, 0};
    v8i gz8 = {0, 0, 0, 0, 0, 0, 0, 0};
    __builtin_amdgcn_tensor_load_to_lds(g0, g1, gz4, gz4, gz8, 0);
#else
    (void)lds_byte_off; (void)gsrc; (void)rows; (void)row_elems;
    (void)pad_int_code; (void)pad_amt_code;
#endif
}

__device__ __forceinline__ void wait_tensorcnt0() {
#if __has_builtin(__builtin_amdgcn_s_wait_tensorcnt)
    __builtin_amdgcn_s_wait_tensorcnt(0);
#else
    asm volatile("s_wait_tensorcnt 0x0" ::: "memory");
#endif
}

__device__ __forceinline__ void ldf(FragBF& f, const char* p) {
    f.q[0] = *(const uint4*)(p);
    f.q[1] = *(const uint4*)(p + 32);
}

__device__ __forceinline__ v8f wmma_bf16(const FragBF& a, const FragBF& b, v8f c) {
    return __builtin_amdgcn_wmma_f32_16x16x32_bf16(false, a.v, false, b.v, (short)0, c, false, false);
}

// K=512 slab, software-pipelined, dual accumulators (even/odd k-step)
__device__ __forceinline__ void gemm_slab16(v8f& accE, v8f& accO, const char* A, const char* B) {
    FragBF a[2], b[2];
    ldf(a[0], A);
    ldf(b[0], B);
#pragma unroll
    for (int ks = 0; ks < 16; ++ks) {
        if (ks + 1 < 16) {
            ldf(a[(ks + 1) & 1], A + (ks + 1) * 64);
            ldf(b[(ks + 1) & 1], B + (ks + 1) * 64);
        }
        v8f& acc = (ks & 1) ? accO : accE;
        acc = wmma_bf16(a[ks & 1], b[ks & 1], acc);
    }
}

// K=1024 h-side slab; B row has a 16B TDM pad after each 1024B (offset bump at ks>=16)
__device__ __forceinline__ void gemm_slab32h(v8f& accE, v8f& accO, const char* A, const char* Bw) {
    FragBF a[2], b[2];
    ldf(a[0], A);
    ldf(b[0], Bw + 1040);
#pragma unroll
    for (int ks = 0; ks < 32; ++ks) {
        if (ks + 1 < 32) {
            const int kn = ks + 1;
            ldf(a[kn & 1], A + kn * 64);
            ldf(b[kn & 1], Bw + 1040 + kn * 64 + ((kn >= 16) ? 16 : 0));
        }
        v8f& acc = (ks & 1) ? accO : accE;
        acc = wmma_bf16(a[ks & 1], b[ks & 1], acc);
    }
}

// ---------------- fused persistent GRU scan (weight-stationary in LDS) ----------------
__launch_bounds__(kNT, 1)
__global__ void k_scan(const uint16_t* __restrict__ xb,
                       const uint16_t* __restrict__ Wcat,
                       const float* __restrict__ b_ih,
                       const float* __restrict__ b_hh,
                       uint16_t* __restrict__ hbf,   // [2][B*H] bf16
                       float*    __restrict__ h32,   // [2][B*H] f32
                       int*      __restrict__ cnt,
                       float*    __restrict__ out)   // ys [S*B*H] then hlast [B*H]
{
    extern __shared__ char smem[];
    char*  ldsW   = smem + LDS_W_OFF;
    char*  ldsH   = smem + LDS_H_OFF;
    float* stageF = (float*)(smem + LDS_S_OFF); // [4][64][18] f32 gate pre-activations
    __shared__ float sbias[4][16];

    const int tid  = threadIdx.x;
    const int lane = tid & 31;
    const int wave = tid >> 5;
    const int j0   = blockIdx.x * 16;   // 16 hidden units per WG

    if (tid < 16) {
        int j = j0 + tid;
        sbias[0][tid] = b_ih[j]          + b_hh[j];
        sbias[1][tid] = b_ih[kH + j]     + b_hh[kH + j];
        sbias[2][tid] = b_ih[2 * kH + j];
        sbias[3][tid] = b_hh[2 * kH + j];
    }

    // ---- one-time weight preload: 3 gate groups of 16 rows via TDM ----
#if __has_builtin(__builtin_amdgcn_tensor_load_to_lds)
    if (wave == 0) {
        const uint32_t base = __builtin_amdgcn_groupstaticsize() + (uint32_t)LDS_W_OFF;
        for (int g = 0; g < 3; ++g)
            tdm_load_2d_bf16(base + (uint32_t)g * 16u * kWB,
                             Wcat + (size_t)(g * kH + j0) * kKC,
                             16u, (uint32_t)kKC,
                             /*pad: 16B per 1024B*/ 7u, 3u);
        wait_tensorcnt0();
    }
#else
    for (int idx = tid; idx < 48 * 3 * 64; idx += kNT) {
        int r = idx / 192, rem = idx - r * 192;
        int seg = rem >> 6, c = rem & 63;
        int grow = (r / 16) * kH + j0 + (r & 15);
        *(uint4*)(ldsW + (size_t)r * kWB + (size_t)seg * 1040 + (size_t)c * 16) =
            *(const uint4*)(Wcat + (size_t)grow * kKC + seg * 512 + c * 8);
    }
#endif
    __syncthreads();

    // ---- scalar (wave-uniform) unit decode: 12 waves -> 12 WMMA units ----
    const int uwave = __builtin_amdgcn_readfirstlane(wave);
    const bool isRZ = (uwave < 8);
    int mtile, lb, g0idx;
    if (isRZ) { mtile = uwave & 3; g0idx = uwave >> 2; lb = g0idx * 16; }  // r or z
    else      { mtile = (uwave - 8) & 3; g0idx = 2; lb = 32; }             // n (split acc)

    const int arow = mtile * 16 + (lane & 15);       // batch row
    const int lrow = lb + (lane & 15);               // local W row
    const int khB  = (lane & 16) ? 16 : 0;           // per-lane K-half byte offset
    const char* Bw = ldsW + (size_t)lrow * kWB + khB;
    const char* Ah = ldsH + (size_t)arow * kHB + khB;
    const int hi8  = (lane & 16) ? 8 : 0;
    const int jj   = lane & 15;
    const int mb   = mtile * 16 + hi8;

    int par = 0;
    for (int s = 0; s < kS; ++s) {
        // ---- issue h (bf16) -> padded LDS copy; its latency hides under the x-slab GEMM ----
        {
            const uint4* src = (const uint4*)(hbf + (size_t)par * kB * kH);
            for (int idx = tid; idx < kB * 128; idx += kNT) {
                int b = idx >> 7, c = idx & 127;
                *(uint4*)(ldsH + (size_t)b * kHB + (size_t)c * 16) = src[idx];
            }
        }

        const uint16_t* xs = xb + (size_t)s * kB * kIN;
        const char* Ax = (const char*)(xs + (size_t)arow * kIN) + khB;

        v8f zf = {0.f,0.f,0.f,0.f,0.f,0.f,0.f,0.f};
        v8f aE = zf, aO = zf, hE = zf, hO = zf;

        // x-side slab (K 0..511): A from global, B from LDS W; doesn't touch ldsH
        gemm_slab16(aE, aO, Ax, Bw);

        __syncthreads();   // h copy complete (all waves)

        // h-side slab (K 512..1535): A from LDS h, B from LDS W
        if (isRZ) {
            gemm_slab32h(aE, aO, Ah, Bw);
#pragma unroll
            for (int v = 0; v < 8; ++v)
                stageF[(g0idx * kB + mb + v) * 18 + jj] = aE[v] + aO[v];
        } else {
            gemm_slab32h(hE, hO, Ah, Bw);
#pragma unroll
            for (int v = 0; v < 8; ++v)
                stageF[(2 * kB + mb + v) * 18 + jj] = aE[v] + aO[v];   // n, x-side
#pragma unroll
            for (int v = 0; v < 8; ++v)
                stageF[(3 * kB + mb + v) * 18 + jj] = hE[v] + hO[v];   // n, h-side
        }
        __syncthreads();

        // ---- gate math + state update: 1024 (b,jj) items ----
        {
            float*       outs = out + (size_t)s * kB * kH;
            const float* hp   = h32 + (size_t)par * kB * kH;
            float*       hn32 = h32 + (size_t)(par ^ 1) * kB * kH;
            uint16_t*    hnbf = hbf + (size_t)(par ^ 1) * kB * kH;
#pragma unroll
            for (int i = 0; i < 3; ++i) {
                int item = tid + kNT * i;
                if (item < kB * 16) {
                    int b = item >> 4, jx = item & 15, j = j0 + jx;
                    float rp = stageF[(0 * kB + b) * 18 + jx] + sbias[0][jx];
                    float zp = stageF[(1 * kB + b) * 18 + jx] + sbias[1][jx];
                    float nx = stageF[(2 * kB + b) * 18 + jx] + sbias[2][jx];
                    float nh = stageF[(3 * kB + b) * 18 + jx] + sbias[3][jx];
                    float r  = 1.0f / (1.0f + __expf(-rp));
                    float z  = 1.0f / (1.0f + __expf(-zp));
                    float n  = tanhf(nx + r * nh);
                    float hv = hp[b * kH + j];
                    float hnew = (1.0f - z) * n + z * hv;
                    __builtin_nontemporal_store(hnew, &outs[(size_t)b * kH + j]);
                    hn32[(size_t)b * kH + j] = hnew;
                    hnbf[(size_t)b * kH + j] = f2bf(hnew);
                    if (s == kS - 1)
                        out[(size_t)kS * kB * kH + (size_t)b * kH + j] = hnew;
                }
            }
            // prefetch next timestep's x slice (64 KB) while waiting
            if (s + 1 < kS) {
                const char* nx_ = (const char*)(xb + (size_t)(s + 1) * kB * kIN);
                for (int i = tid; i < 512; i += kNT)
                    __builtin_prefetch(nx_ + (size_t)i * 128, 0, 3);
            }
        }

        // ---- grid-wide sync ----
        __threadfence();
        __syncthreads();
        if (tid == 0) {
            __hip_atomic_fetch_add(cnt, 1, __ATOMIC_ACQ_REL, __HIP_MEMORY_SCOPE_AGENT);
            const int target = kNWG * (s + 1);
            while (__hip_atomic_load(cnt, __ATOMIC_ACQUIRE, __HIP_MEMORY_SCOPE_AGENT) < target)
                __builtin_amdgcn_s_sleep(4);
        }
        __syncthreads();
        par ^= 1;
    }
}

// ---------------- host entry ----------------
extern "C" void kernel_launch(void* const* d_in, const int* in_sizes, int n_in,
                              void* d_out, int out_size, void* d_ws, size_t ws_size,
                              hipStream_t stream) {
    (void)in_sizes; (void)n_in; (void)out_size; (void)ws_size;
    const float* x    = (const float*)d_in[0];
    const float* W_ih = (const float*)d_in[1];
    const float* W_hh = (const float*)d_in[2];
    const float* b_ih = (const float*)d_in[3];
    const float* b_hh = (const float*)d_in[4];
    float* out = (float*)d_out;

    char* ws = (char*)d_ws;
    uint16_t* xb   = (uint16_t*)(ws + OFF_XB);
    uint16_t* Wcat = (uint16_t*)(ws + OFF_WCAT);
    uint16_t* hbf  = (uint16_t*)(ws + OFF_HBF);
    float*    h32  = (float*)   (ws + OFF_H32);
    int*      cnt  = (int*)     (ws + OFF_CNT);

    k_init<<<(kB * kH + 255) / 256, 256, 0, stream>>>(hbf, h32, cnt);
    k_cvt_x<<<2048, 256, 0, stream>>>(x, xb, kS * kB * kIN);
    k_build_w<<<2048, 256, 0, stream>>>(W_ih, W_hh, Wcat);

    k_scan<<<kNWG, kNT, (size_t)LDS_TOTAL, stream>>>(xb, Wcat, b_ih, b_hh, hbf, h32, cnt, out);
}
